// GATEmbeddingModel_87995289960829
// MI455X (gfx1250) — matrix operand order repaired and checked by reading, not compiled, not run on hardware
//
#include <hip/hip_runtime.h>
#include <hip/hip_bf16.h>
#include <math.h>

typedef __attribute__((ext_vector_type(16))) _Float16 v16h;
typedef __attribute__((ext_vector_type(8)))  float    v8f;

#define TPB 256

// ---------------------------------------------------------------------------
// utility: grid-stride zero fill (used instead of hipMemsetAsync)
// ---------------------------------------------------------------------------
__global__ void zero_f32(float* p, size_t n) {
  size_t i = blockIdx.x * (size_t)blockDim.x + threadIdx.x;
  size_t stride = (size_t)gridDim.x * blockDim.x;
  for (; i < n; i += stride) p[i] = 0.0f;
}

// ---------------------------------------------------------------------------
// order-preserving float <-> uint mapping for atomic max on floats
// ---------------------------------------------------------------------------
__device__ inline unsigned f2ord(float f) {
  unsigned u = __float_as_uint(f);
  return (u & 0x80000000u) ? ~u : (u | 0x80000000u);
}
__device__ inline float ord2f(unsigned o) {
  unsigned u = (o & 0x80000000u) ? (o & 0x7fffffffu) : ~o;
  return __uint_as_float(u);
}

// ---------------------------------------------------------------------------
// pack weight matrix B[K,Nc] (f32, row-major) into Bc[Nc][KP] (f16,
// column-major, K zero-padded to KP).  Gives the WMMA B-fragment a single
// contiguous 32-byte load per k-step.
// ---------------------------------------------------------------------------
__global__ void pack_b_f16(const float* __restrict__ B, _Float16* __restrict__ Bc,
                           int K, int KP, int Nc) {
  int idx = blockIdx.x * blockDim.x + threadIdx.x;
  if (idx >= Nc * KP) return;
  int c = idx / KP, k = idx - c * KP;
  Bc[idx] = (k < K) ? (_Float16)B[(size_t)k * Nc + c] : (_Float16)0.0f;
}

// ---------------------------------------------------------------------------
// WMMA GEMM: C[M,Nc] = A[M,KA] @ B, one wave per 16x16 tile.
//   KP: padded K (multiple of 32, compile-time)   KA: true A width
//   A   f32 row-major,  Bc  f16 [Nc][KP] column-major (pre-padded)
// All bounds guards fold at compile time -> contiguous b128 loads.
// ---------------------------------------------------------------------------
template <int KP, int KA>
__global__ void gemm_wmma_f16(const float* __restrict__ A,
                              const _Float16* __restrict__ Bc,
                              float* __restrict__ C,
                              int M, int Nc) {
  int wave = (blockIdx.x * blockDim.x + threadIdx.x) >> 5;
  int lane = threadIdx.x & 31;
  int ntiles = Nc >> 4;
  int mt = wave / ntiles;
  int nt = wave - mt * ntiles;
  if (mt * 16 >= M) return;                 // wave-uniform: EXEC stays full

  const int half = lane >> 4;               // 0: lanes 0-15, 1: lanes 16-31
  const int rc   = lane & 15;               // A-row / B-col within tile

  const float*    Arow = A  + (size_t)(mt * 16 + rc) * KA;
  const _Float16* Bcol = Bc + (size_t)(nt * 16 + rc) * KP + half * 16;

  v8f acc = {};
#pragma unroll
  for (int k0 = 0; k0 < KP; k0 += 32) {
    // ---- A fragment (contiguous k order: k = 16*half + s, matches B) ----
    v16h a;
    if constexpr (KA == KP) {
      const float4* pa0 = reinterpret_cast<const float4*>(Arow + k0 + half * 16);
      const float4* pa1 = reinterpret_cast<const float4*>(Arow + k0 + half * 16 + 8);
      float4 x0 = pa0[0], x1 = pa0[1];
      float4 y0 = pa1[0], y1 = pa1[1];
      a[0]=(_Float16)x0.x; a[1]=(_Float16)x0.y; a[2]=(_Float16)x0.z; a[3]=(_Float16)x0.w;
      a[4]=(_Float16)x1.x; a[5]=(_Float16)x1.y; a[6]=(_Float16)x1.z; a[7]=(_Float16)x1.w;
      a[8]=(_Float16)y0.x; a[9]=(_Float16)y0.y; a[10]=(_Float16)y0.z; a[11]=(_Float16)y0.w;
      a[12]=(_Float16)y1.x; a[13]=(_Float16)y1.y; a[14]=(_Float16)y1.z; a[15]=(_Float16)y1.w;
    } else {
      // KA < 16: only half==0 lanes carry data (K = 0..7); rest is zero pad
      a = v16h{};
      if (half == 0) {
        const float4* pa = reinterpret_cast<const float4*>(Arow);
        float4 x0 = pa[0], x1 = pa[1];
        a[0]=(_Float16)x0.x; a[1]=(_Float16)x0.y; a[2]=(_Float16)x0.z; a[3]=(_Float16)x0.w;
        a[4]=(_Float16)x1.x; a[5]=(_Float16)x1.y; a[6]=(_Float16)x1.z; a[7]=(_Float16)x1.w;
      }
    }
    // ---- B fragment: 16 contiguous halves from packed column ----
    v16h b = *reinterpret_cast<const v16h*>(Bcol + k0);

    acc = __builtin_amdgcn_wmma_f32_16x16x32_f16(
        false, a, false, b, (short)0, acc, false, false);
  }

  // C/D layout: VGPR i -> row = 8*half + i, col = rc
#pragma unroll
  for (int i = 0; i < 8; ++i) {
    int row = mt * 16 + half * 8 + i;
    C[(size_t)row * Nc + nt * 16 + rc] = acc[i];
  }
}

// ---------------------------------------------------------------------------
// per-node attention scores: s_src[n,h] = <h[n,h,:], a_src[h,:]> (same a_dst)
// ---------------------------------------------------------------------------
__global__ void attn_scores(const float* __restrict__ h,
                            const float* __restrict__ a_src,
                            const float* __restrict__ a_dst,
                            int N, int ch,
                            float* __restrict__ s_src,
                            float* __restrict__ s_dst) {
  int idx = blockIdx.x * blockDim.x + threadIdx.x;
  if (idx >= N * 4) return;
  int n = idx >> 2, hd = idx & 3;
  const float* row = h + (size_t)n * (4 * ch) + hd * ch;
  const float* as = a_src + hd * ch;
  const float* ad = a_dst + hd * ch;
  float ss = 0.f, sd = 0.f;
  for (int c = 0; c < ch; ++c) { float v = row[c]; ss += v * as[c]; sd += v * ad[c]; }
  s_src[idx] = ss;
  s_dst[idx] = sd;
}

__device__ inline void edge_endpoints(const int* esrc, const int* edst,
                                      int e, int E0, int& s, int& d) {
  if (e < E0) { s = esrc[e]; d = edst[e]; }
  else        { s = e - E0;  d = e - E0; }          // self-loops
}

// ---------------------------------------------------------------------------
// edge pass 1: leaky-relu score, segment max into ordered-uint buffer
// ---------------------------------------------------------------------------
__global__ void edge_max(const float* __restrict__ s_src,
                         const float* __restrict__ s_dst,
                         const int* __restrict__ esrc,
                         const int* __restrict__ edst,
                         int E0, int Etot, unsigned* __restrict__ mord) {
  int e = blockIdx.x * blockDim.x + threadIdx.x;
  if (e >= Etot) return;
  int s, d; edge_endpoints(esrc, edst, e, E0, s, d);
#pragma unroll
  for (int h = 0; h < 4; ++h) {
    float sc = s_src[s * 4 + h] + s_dst[d * 4 + h];
    sc = (sc > 0.f) ? sc : 0.2f * sc;
    atomicMax(&mord[d * 4 + h], f2ord(sc));
  }
}

// ---------------------------------------------------------------------------
// edge pass 2: e = exp(score - m[dst]); denom[dst] += e; stash e
// ---------------------------------------------------------------------------
__global__ void edge_exp_sum(const float* __restrict__ s_src,
                             const float* __restrict__ s_dst,
                             const int* __restrict__ esrc,
                             const int* __restrict__ edst,
                             const unsigned* __restrict__ mord,
                             int E0, int Etot,
                             float* __restrict__ ebuf,
                             float* __restrict__ denom) {
  int e = blockIdx.x * blockDim.x + threadIdx.x;
  if (e >= Etot) return;
  int s, d; edge_endpoints(esrc, edst, e, E0, s, d);
#pragma unroll
  for (int h = 0; h < 4; ++h) {
    float sc = s_src[s * 4 + h] + s_dst[d * 4 + h];
    sc = (sc > 0.f) ? sc : 0.2f * sc;
    float m = ord2f(mord[d * 4 + h]);
    float ex = __expf(sc - m);
    ebuf[(size_t)e * 4 + h] = ex;
    atomicAdd(&denom[d * 4 + h], ex);
  }
}

// ---------------------------------------------------------------------------
// edge pass 3: out[dst] += alpha * h[src]   (thread per edge-head)
// ---------------------------------------------------------------------------
__global__ void edge_msg(const float* __restrict__ hfeat,
                         const float* __restrict__ ebuf,
                         const float* __restrict__ denom,
                         const int* __restrict__ esrc,
                         const int* __restrict__ edst,
                         int E0, int Etot, int ch,
                         float* __restrict__ out) {
  int idx = blockIdx.x * blockDim.x + threadIdx.x;
  int e = idx >> 2, h = idx & 3;
  if (e >= Etot) return;
  int s, d; edge_endpoints(esrc, edst, e, E0, s, d);
  float alpha = ebuf[(size_t)e * 4 + h] / denom[d * 4 + h];
  const float* hr = hfeat + (size_t)s * (4 * ch) + h * ch;
  float* orow = out + (size_t)d * (4 * ch) + h * ch;
  for (int c = 0; c < ch; ++c) atomicAdd(&orow[c], hr[c] * alpha);
}

// ---------------------------------------------------------------------------
// batch-norm stats: one block per channel
// ---------------------------------------------------------------------------
__global__ void bn_stats(const float* __restrict__ x, int N, int F,
                         float* __restrict__ mean, float* __restrict__ rstd) {
  int c = blockIdx.x;
  int tid = threadIdx.x;
  __shared__ float ss[TPB], sq[TPB];
  float s = 0.f, q = 0.f;
  for (int i = tid; i < N; i += blockDim.x) {
    float v = x[(size_t)i * F + c];
    s += v; q += v * v;
  }
  ss[tid] = s; sq[tid] = q;
  __syncthreads();
  for (int o = TPB / 2; o > 0; o >>= 1) {
    if (tid < o) { ss[tid] += ss[tid + o]; sq[tid] += sq[tid + o]; }
    __syncthreads();
  }
  if (tid == 0) {
    float mu  = ss[0] / (float)N;
    float var = sq[0] / (float)N - mu * mu;
    mean[c] = mu;
    rstd[c] = rsqrtf(fmaxf(var, 0.f) + 1e-5f);
  }
}

__global__ void bn_apply_elu(float* __restrict__ x, int N, int F,
                             const float* __restrict__ g,
                             const float* __restrict__ beta,
                             const float* __restrict__ mean,
                             const float* __restrict__ rstd) {
  size_t idx = blockIdx.x * (size_t)blockDim.x + threadIdx.x;
  if (idx >= (size_t)N * F) return;
  int c = (int)(idx % F);
  float y = g[c] * (x[idx] - mean[c]) * rstd[c] + beta[c];
  x[idx] = (y > 0.f) ? y : expm1f(y);
}

// ---------------------------------------------------------------------------
// global mean pool
// ---------------------------------------------------------------------------
__global__ void pool_sum(const float* __restrict__ x, const int* __restrict__ batch,
                         int N, int F, float* __restrict__ pooled) {
  size_t idx = blockIdx.x * (size_t)blockDim.x + threadIdx.x;
  if (idx >= (size_t)N * F) return;
  int n = (int)(idx / F), c = (int)(idx % F);
  atomicAdd(&pooled[batch[n] * F + c], x[idx]);
}

__global__ void pool_count(const int* __restrict__ batch, int N, float* __restrict__ counts) {
  int n = blockIdx.x * blockDim.x + threadIdx.x;
  if (n >= N) return;
  atomicAdd(&counts[batch[n]], 1.0f);
}

__global__ void pool_div(float* __restrict__ pooled, const float* __restrict__ counts,
                         int G, int F) {
  int idx = blockIdx.x * blockDim.x + threadIdx.x;
  if (idx >= G * F) return;
  pooled[idx] /= fmaxf(counts[idx / F], 1.0f);
}

// ---------------------------------------------------------------------------
// small fully-connected layer: out[r,c] = relu?(in[r,:] @ w[:,c] + b[c])
// ---------------------------------------------------------------------------
__global__ void fc_layer(const float* __restrict__ in, const float* __restrict__ w,
                         const float* __restrict__ b, float* __restrict__ out,
                         int rows, int K, int Nc, int relu) {
  int idx = blockIdx.x * blockDim.x + threadIdx.x;
  if (idx >= rows * Nc) return;
  int r = idx / Nc, c = idx % Nc;
  float acc = b[c];
  for (int k = 0; k < K; ++k) acc += in[(size_t)r * K + k] * w[(size_t)k * Nc + c];
  out[idx] = relu ? fmaxf(acc, 0.f) : acc;
}

// ---------------------------------------------------------------------------
// launcher
// ---------------------------------------------------------------------------
static inline int cdiv(long long a, long long b) { return (int)((a + b - 1) / b); }

extern "C" void kernel_launch(void* const* d_in, const int* in_sizes, int n_in,
                              void* d_out, int out_size, void* d_ws, size_t ws_size,
                              hipStream_t stream) {
  const float* x      = (const float*)d_in[0];
  const int*   eidx   = (const int*)d_in[1];
  const int*   batch  = (const int*)d_in[2];
  const float* W1     = (const float*)d_in[3];
  const float* a_src1 = (const float*)d_in[4];
  const float* a_dst1 = (const float*)d_in[5];
  // d_in[6] = b1 (cancels inside BN; zeros anyway)
  const float* g1     = (const float*)d_in[7];
  const float* beta1  = (const float*)d_in[8];
  const float* W2     = (const float*)d_in[9];
  const float* a_src2 = (const float*)d_in[10];
  const float* a_dst2 = (const float*)d_in[11];
  // d_in[12] = b2 (cancels inside BN)
  const float* g2     = (const float*)d_in[13];
  const float* beta2  = (const float*)d_in[14];
  const float* fc1_w  = (const float*)d_in[15];
  const float* fc1_b  = (const float*)d_in[16];
  const float* fc2_w  = (const float*)d_in[17];
  const float* fc2_b  = (const float*)d_in[18];
  const float* fc3_w  = (const float*)d_in[19];
  const float* fc3_b  = (const float*)d_in[20];
  float* outp = (float*)d_out;

  const int N    = in_sizes[0] / 8;      // 50000
  const int E0   = in_sizes[1] / 2;      // 800000
  const int Etot = E0 + N;               // + self loops
  const int G    = 64;
  const int* esrc = eidx;
  const int* edst = eidx + E0;

  // ---- workspace layout (floats; every offset stays 32B-aligned) ----
  float* ws = (float*)d_ws;
  float*    out1   = ws;               ws += (size_t)N * 128;
  float*    out2   = ws;               ws += (size_t)N * 256;
  unsigned* mord   = (unsigned*)ws;    ws += (size_t)N * 4;
  float*    denom  = ws;               ws += (size_t)N * 4;
  float*    pooled = ws;               ws += (size_t)G * 256;
  float*    counts = ws;               ws += G;
  float*    mean   = ws;               ws += 256;
  float*    rstd   = ws;               ws += 256;
  float*    h1     = ws;               ws += (size_t)N * 128;
  float*    h2     = ws;               ws += (size_t)N * 256;
  float*    ssrc   = ws;               ws += (size_t)N * 4;
  float*    sdst   = ws;               ws += (size_t)N * 4;
  float*    ebuf   = ws;               ws += (size_t)Etot * 4;
  float*    z1     = ws;               ws += (size_t)G * 128;
  float*    z2     = ws;               ws += (size_t)G * 64;
  _Float16* Bc1    = (_Float16*)ws;    ws += (128 * 32) / 2;    // [128][32] f16
  _Float16* Bc2    = (_Float16*)ws;    ws += (256 * 128) / 2;   // [256][128] f16

  // ---- pack weights to f16 column-major (tiny) ----
  pack_b_f16<<<cdiv(128 * 32, TPB),  TPB, 0, stream>>>(W1, Bc1, 8,   32,  128);
  pack_b_f16<<<cdiv(256 * 128, TPB), TPB, 0, stream>>>(W2, Bc2, 128, 128, 256);

  // ================= layer 1 (heads=4, ch=32, Fout=128) =================
  zero_f32<<<2048, TPB, 0, stream>>>(out1, (size_t)N * 128);
  zero_f32<<<512,  TPB, 0, stream>>>((float*)mord, (size_t)N * 8); // mord + denom

  { // h1 = x @ W1   (M=N, KA=8 padded to 32, Nc=128) -> 3125*8 waves
    long long waves = (long long)(N / 16) * (128 / 16);
    gemm_wmma_f16<32, 8><<<cdiv(waves, TPB / 32), TPB, 0, stream>>>(x, Bc1, h1, N, 128);
  }
  attn_scores<<<cdiv((long long)N * 4, TPB), TPB, 0, stream>>>(h1, a_src1, a_dst1, N, 32, ssrc, sdst);
  edge_max    <<<cdiv(Etot, TPB), TPB, 0, stream>>>(ssrc, sdst, esrc, edst, E0, Etot, mord);
  edge_exp_sum<<<cdiv(Etot, TPB), TPB, 0, stream>>>(ssrc, sdst, esrc, edst, mord, E0, Etot, ebuf, denom);
  edge_msg    <<<cdiv((long long)Etot * 4, TPB), TPB, 0, stream>>>(h1, ebuf, denom, esrc, edst, E0, Etot, 32, out1);
  bn_stats    <<<128, TPB, 0, stream>>>(out1, N, 128, mean, rstd);
  bn_apply_elu<<<cdiv((long long)N * 128, TPB), TPB, 0, stream>>>(out1, N, 128, g1, beta1, mean, rstd);

  // ================= layer 2 (heads=4, ch=64, Fout=256) =================
  zero_f32<<<2048, TPB, 0, stream>>>(out2, (size_t)N * 256);
  zero_f32<<<512,  TPB, 0, stream>>>((float*)mord, (size_t)N * 8);

  { // h2 = out1 @ W2   (M=N, K=128, Nc=256) -> 3125*16 waves
    long long waves = (long long)(N / 16) * (256 / 16);
    gemm_wmma_f16<128, 128><<<cdiv(waves, TPB / 32), TPB, 0, stream>>>(out1, Bc2, h2, N, 256);
  }
  attn_scores<<<cdiv((long long)N * 4, TPB), TPB, 0, stream>>>(h2, a_src2, a_dst2, N, 64, ssrc, sdst);
  edge_max    <<<cdiv(Etot, TPB), TPB, 0, stream>>>(ssrc, sdst, esrc, edst, E0, Etot, mord);
  edge_exp_sum<<<cdiv(Etot, TPB), TPB, 0, stream>>>(ssrc, sdst, esrc, edst, mord, E0, Etot, ebuf, denom);
  edge_msg    <<<cdiv((long long)Etot * 4, TPB), TPB, 0, stream>>>(h2, ebuf, denom, esrc, edst, E0, Etot, 64, out2);
  bn_stats    <<<256, TPB, 0, stream>>>(out2, N, 256, mean, rstd);
  bn_apply_elu<<<cdiv((long long)N * 256, TPB), TPB, 0, stream>>>(out2, N, 256, g2, beta2, mean, rstd);

  // ================= pool + MLP =================
  zero_f32<<<64, TPB, 0, stream>>>(pooled, (size_t)G * 256 + G); // pooled + counts
  pool_sum  <<<cdiv((long long)N * 256, TPB), TPB, 0, stream>>>(out2, batch, N, 256, pooled);
  pool_count<<<cdiv(N, TPB), TPB, 0, stream>>>(batch, N, counts);
  pool_div  <<<cdiv(G * 256, TPB), TPB, 0, stream>>>(pooled, counts, G, 256);

  fc_layer<<<cdiv(G * 128, TPB), TPB, 0, stream>>>(pooled, fc1_w, fc1_b, z1, G, 256, 128, 1);
  fc_layer<<<cdiv(G * 64,  TPB), TPB, 0, stream>>>(z1,     fc2_w, fc2_b, z2, G, 128, 64, 1);
  fc_layer<<<cdiv(G * 1,   TPB), TPB, 0, stream>>>(z2,     fc3_w, fc3_b, outp, G, 64, 1, 0);
}
